// InstantLoRAMidModule_72206990180685
// MI455X (gfx1250) — matrix operand order repaired and checked by reading, not compiled, not run on hardware
//
#include <hip/hip_runtime.h>
#include <hip/hip_bf16.h>
#include <stdint.h>

#define B_  8
#define S_  4096
#define D_  1280
#define R_  64
#define EW_ (D_ + R_*R_ + D_)   // 6656

typedef __bf16          v16bf __attribute__((ext_vector_type(16)));
typedef __bf16          v8bf  __attribute__((ext_vector_type(8)));
typedef float           v8f   __attribute__((ext_vector_type(8)));
typedef float           f4v   __attribute__((ext_vector_type(4)));
typedef unsigned short  v16u  __attribute__((ext_vector_type(16)));

// ---------------------------------------------------------------------------
// Prep 1: W1[b][r][d] = w_down[r][d] * down_w[b][d], stored as bf16 in the
// exact WMMA B-fragment layout for the stage-1 (K=D, N=R) GEMM:
//   [b][kt(40)][nt(4)][lane(32)][16]  (32B contiguous per lane)
// ---------------------------------------------------------------------------
__global__ void prep_w1(const float* __restrict__ wdown,
                        const float* __restrict__ embed,
                        __bf16* __restrict__ w1f) {
  int tid = blockIdx.x * blockDim.x + threadIdx.x;
  if (tid >= B_ * D_ * R_) return;
  int n = tid & (R_ - 1);
  int k = (tid >> 6) % D_;
  int b = tid / (R_ * D_);
  float v = wdown[n * D_ + k] * embed[b * EW_ + k];
  int kt = k >> 5, kl = k & 31;
  int lane = (n & 15) | (((kl >> 3) & 1) << 4);
  int j = (kl & 7) | ((kl >> 4) << 3);
  int idx = ((((b * 40 + kt) * 4 + (n >> 4)) * 32 + lane) << 4) + j;
  w1f[idx] = (__bf16)v;
}

// ---------------------------------------------------------------------------
// Prep 2: W2[b][d][r] = up_w[b][d] * sum_q w_up[d][q] * mid_w[b][q][r]
// stored in WMMA B-fragment layout for stage-2 (K=R, N=D):
//   [b][kt(2)][nt(80)][lane(32)][16]
// bias[b][d] = up_w*b_up[d] + sum_r b_down[r]*W2[b][d][r]
// ---------------------------------------------------------------------------
__global__ void prep_w2(const float* __restrict__ wup,
                        const float* __restrict__ embed,
                        const float* __restrict__ bdown,
                        const float* __restrict__ bup,
                        __bf16* __restrict__ w2f,
                        float* __restrict__ biasb) {
  int tid = blockIdx.x * blockDim.x + threadIdx.x;
  if (tid >= B_ * D_) return;
  int d = tid % D_;
  int b = tid / D_;
  const float* mid = embed + (size_t)b * EW_ + D_;
  const float* wu  = wup + (size_t)d * R_;
  float up = embed[(size_t)b * EW_ + D_ + R_ * R_ + d];
  float tmp[R_];
  float bias = up * bup[d];
  #pragma unroll 4
  for (int r = 0; r < R_; ++r) {
    float acc = 0.f;
    for (int q = 0; q < R_; ++q) acc += wu[q] * mid[q * R_ + r];
    float w = up * acc;
    tmp[r] = w;
    bias += bdown[r] * w;
  }
  biasb[b * D_ + d] = bias;
  int nt = d >> 4;
  for (int r = 0; r < R_; ++r) {
    int kt = r >> 5, kl = r & 31;
    int lane = (d & 15) | (((kl >> 3) & 1) << 4);
    int j = (kl & 7) | ((kl >> 4) << 3);
    int idx = ((((b * 2 + kt) * 80 + nt) * 32 + lane) << 4) + j;
    w2f[idx] = (__bf16)tmp[r];
  }
}

// ---------------------------------------------------------------------------
// Main kernel: 4 waves/block, each wave owns 32 consecutive s-rows.
// Stage 1: h1(32x64) = x(32x1280) @ W1^T via 40 K-chunks * 4 N-tiles * 2 M-subs
// Stage 2: out(32x1280) = h1 @ W2^T + bias via 80 N-tiles * 2 K-chunks * 2 M-subs
// ---------------------------------------------------------------------------
__global__ void __launch_bounds__(128) lora_main(
    const float* __restrict__ x,
    const __bf16* __restrict__ w1f,
    const __bf16* __restrict__ w2f,
    const float* __restrict__ biasb,
    float* __restrict__ out) {
  __shared__ __bf16 h1s[4][32][R_];          // 16 KB: per-wave 32x64 bf16

  const int lane = threadIdx.x & 31;
  const int wave = threadIdx.x >> 5;
  const int b     = blockIdx.x >> 5;          // 32 s-tiles per batch
  const int stile = blockIdx.x & 31;
  const int s0   = stile * 128 + wave * 32;
  const int mrow = lane & 15;                 // row within 16-row A tile
  const int kh   = lane >> 4;                 // K-half select

  const float* xrow0 = x + ((size_t)b * S_ + s0 + mrow) * D_;
  const float* xrow1 = xrow0 + (size_t)16 * D_;

  v8f acc1[2][4];
  #pragma unroll
  for (int s = 0; s < 2; ++s)
    #pragma unroll
    for (int n = 0; n < 4; ++n)
      acc1[s][n] = (v8f){0.f, 0.f, 0.f, 0.f, 0.f, 0.f, 0.f, 0.f};

  const v16bf* w1p = ((const v16bf*)w1f) + (size_t)b * 40 * 4 * 32;

  // ---------------- Stage 1: K loop over D ----------------
  for (int kt = 0; kt < 40; ++kt) {
    const int kb = kt * 32 + kh * 8;
    f4v p0 = *(const f4v*)(xrow0 + kb);
    f4v p1 = *(const f4v*)(xrow0 + kb + 4);
    f4v p2 = *(const f4v*)(xrow0 + kb + 16);
    f4v p3 = *(const f4v*)(xrow0 + kb + 20);
    f4v q0 = *(const f4v*)(xrow1 + kb);
    f4v q1 = *(const f4v*)(xrow1 + kb + 4);
    f4v q2 = *(const f4v*)(xrow1 + kb + 16);
    f4v q3 = *(const f4v*)(xrow1 + kb + 20);
    v16bf a0, a1;
    #pragma unroll
    for (int i = 0; i < 4; ++i) {
      a0[i]      = (__bf16)p0[i];  a0[4 + i]  = (__bf16)p1[i];
      a0[8 + i]  = (__bf16)p2[i];  a0[12 + i] = (__bf16)p3[i];
      a1[i]      = (__bf16)q0[i];  a1[4 + i]  = (__bf16)q1[i];
      a1[8 + i]  = (__bf16)q2[i];  a1[12 + i] = (__bf16)q3[i];
    }
    #pragma unroll
    for (int nt = 0; nt < 4; ++nt) {
      v16bf bb = w1p[(kt * 4 + nt) * 32 + lane];
      acc1[0][nt] = __builtin_amdgcn_wmma_f32_16x16x32_bf16(
          false, a0, false, bb, (short)0, acc1[0][nt], false, false);
      acc1[1][nt] = __builtin_amdgcn_wmma_f32_16x16x32_bf16(
          false, a1, false, bb, (short)0, acc1[1][nt], false, false);
    }
  }

  // ---------------- h1 C-layout -> LDS (bf16) ----------------
  #pragma unroll
  for (int s = 0; s < 2; ++s)
    #pragma unroll
    for (int nt = 0; nt < 4; ++nt) {
      v8f c = acc1[s][nt];
      #pragma unroll
      for (int i = 0; i < 8; ++i) {
        int row = s * 16 + kh * 8 + i;     // C-layout: M = i + 8*(lane>=16)
        int col = nt * 16 + mrow;          //           N = lane%16
        h1s[wave][row][col] = (__bf16)c[i];
      }
    }
  __syncthreads();

  // ---------------- Stage 2 A-fragments from LDS ----------------
  v16bf a2[2][2];
  #pragma unroll
  for (int s = 0; s < 2; ++s)
    #pragma unroll
    for (int kt = 0; kt < 2; ++kt) {
      int row = s * 16 + mrow;
      int kb = kt * 32 + kh * 8;
      v8bf lo = *(const v8bf*)&h1s[wave][row][kb];
      v8bf hi = *(const v8bf*)&h1s[wave][row][kb + 16];
      v16bf t;
      #pragma unroll
      for (int i = 0; i < 8; ++i) { t[i] = lo[i]; t[8 + i] = hi[i]; }
      a2[s][kt] = t;
    }

  const v16bf* w2p = ((const v16bf*)w2f) + (size_t)b * 2 * 80 * 32;
  float* ob = out + ((size_t)b * S_ + s0) * D_;
  const float* bp = biasb + b * D_;

  // ---------------- Stage 2: N loop over D ----------------
  for (int nt2 = 0; nt2 < 80; ++nt2) {
    float bv = bp[nt2 * 16 + mrow];
    v8f c0 = (v8f){bv, bv, bv, bv, bv, bv, bv, bv};
    v8f c1 = c0;
    #pragma unroll
    for (int kt = 0; kt < 2; ++kt) {
      v16bf bb = w2p[(kt * 80 + nt2) * 32 + lane];
      c0 = __builtin_amdgcn_wmma_f32_16x16x32_bf16(
          false, a2[0][kt], false, bb, (short)0, c0, false, false);
      c1 = __builtin_amdgcn_wmma_f32_16x16x32_bf16(
          false, a2[1][kt], false, bb, (short)0, c1, false, false);
    }
    int col = nt2 * 16 + mrow;
    #pragma unroll
    for (int i = 0; i < 8; ++i) {
      int r0 = kh * 8 + i;
      ob[(size_t)r0 * D_ + col]        = c0[i];
      ob[(size_t)(16 + r0) * D_ + col] = c1[i];
    }
  }
}

// ---------------------------------------------------------------------------
extern "C" void kernel_launch(void* const* d_in, const int* in_sizes, int n_in,
                              void* d_out, int out_size, void* d_ws, size_t ws_size,
                              hipStream_t stream) {
  const float* x     = (const float*)d_in[0];
  const float* embed = (const float*)d_in[1];
  const float* wdown = (const float*)d_in[2];
  const float* bdown = (const float*)d_in[3];
  const float* wup   = (const float*)d_in[4];
  const float* bup   = (const float*)d_in[5];
  float* out = (float*)d_out;

  __bf16* w1f = (__bf16*)d_ws;                        // 8*40*4*32*16 bf16
  __bf16* w2f = w1f + (size_t)B_ * D_ * R_;           // 8*2*80*32*16 bf16
  float* biasb = (float*)(w2f + (size_t)B_ * D_ * R_); // 8*1280 f32

  prep_w1<<<(B_ * D_ * R_ + 255) / 256, 256, 0, stream>>>(wdown, embed, w1f);
  prep_w2<<<(B_ * D_ + 255) / 256, 256, 0, stream>>>(wup, embed, bdown, bup, w2f, biasb);
  lora_main<<<B_ * (S_ / 128), 128, 0, stream>>>(x, w1f, w2f, biasb, out);
}
